// EdgeDecoder_15496242004449
// MI455X (gfx1250) — matrix operand order repaired and checked by reading, not compiled, not run on hardware
//
#include <hip/hip_runtime.h>
#include <hip/hip_bf16.h>

typedef __attribute__((ext_vector_type(16))) _Float16 v16h;
typedef __attribute__((ext_vector_type(8)))  _Float16 v8h;
typedef __attribute__((ext_vector_type(8)))  float    v8f;

#define D_IN     256
#define PITCH_H  264                 // f16 elements per LDS row (256 + 8 pad)
#define PITCH_B  (PITCH_H * 2)       // 528 bytes per row
#define LDS_BYTES (256 * PITCH_B)    // 135168 B = 132 KB

// ---- prep: W1 f32 -> f16 (row-major [256][256]) into workspace ----
__global__ void cvt_w1_kernel(const float* __restrict__ W1, _Float16* __restrict__ w1h) {
    int i = blockIdx.x * blockDim.x + threadIdx.x;
    if (i < 256 * 256) w1h[i] = (_Float16)W1[i];
}

// Build the 8 K-step A fragments (f16 16x32 layout) for one 16-edge row set.
// Per lane: chunk d0 = kt*32 + hi*8 -> V0..3, chunk d0+16 -> V4..7.
__device__ __forceinline__ void build_A(const float* __restrict__ zs,
                                        const float* __restrict__ zt,
                                        int hi, v16h* __restrict__ A)
{
    #pragma unroll
    for (int kt = 0; kt < 8; ++kt) {
        const int d0 = kt * 32 + hi * 8;
        float4 a0 = *(const float4*)(zs + d0);
        float4 a1 = *(const float4*)(zs + d0 + 4);
        float4 a2 = *(const float4*)(zs + d0 + 16);
        float4 a3 = *(const float4*)(zs + d0 + 20);
        float4 c0 = *(const float4*)(zt + d0);
        float4 c1 = *(const float4*)(zt + d0 + 4);
        float4 c2 = *(const float4*)(zt + d0 + 16);
        float4 c3 = *(const float4*)(zt + d0 + 20);
        float p[16] = { a0.x*c0.x, a0.y*c0.y, a0.z*c0.z, a0.w*c0.w,
                        a1.x*c1.x, a1.y*c1.y, a1.z*c1.z, a1.w*c1.w,
                        a2.x*c2.x, a2.y*c2.y, a2.z*c2.z, a2.w*c2.w,
                        a3.x*c3.x, a3.y*c3.y, a3.z*c3.z, a3.w*c3.w };
        #pragma unroll
        for (int i = 0; i < 16; ++i) A[kt][i] = (_Float16)p[i];
    }
}

// ---- fused edge MLP, 32 edges per wave (2 C tiles share every B fragment) ----
__launch_bounds__(256, 1)
__global__ void edge_mlp_kernel(const float* __restrict__ z,
                                const long long* __restrict__ edge,
                                const _Float16* __restrict__ w1h,
                                const float* __restrict__ b1,
                                const float* __restrict__ W2,
                                const float* __restrict__ b2,
                                float* __restrict__ out,
                                int E, int ntiles /* 32-edge tiles */)
{
    extern __shared__ char smem[];
    const int tid = threadIdx.x;

    // cooperatively stage f16 W1 into padded LDS (rows = hidden unit n, cols = d)
    for (int c = tid; c < 8192; c += 256) {          // 8192 chunks of 16B
        int row = c >> 5;
        int col = c & 31;
        uint4 v = ((const uint4*)w1h)[(row << 5) + col];
        *(uint4*)(smem + row * PITCH_B + (col << 4)) = v;
    }
    __syncthreads();

    const int lane = tid & 31;
    const int m    = lane & 15;    // A row (edge within row-set) == C column N
    const int hi   = lane >> 4;

    const int gwave  = blockIdx.x * 8 + (tid >> 5);
    const int nwaves = gridDim.x * 8;
    const float b2v  = b2[0];

    for (int tile = gwave; tile < ntiles; tile += nwaves) {
        const long long ebase = (long long)tile * 32;

        // ---- gather + multiply + cvt: two 16-edge A sets, 128 VGPRs total ----
        v16h A[16];
        {
            long long e0 = ebase + m;
            long long e1 = ebase + 16 + m;
            long long c0 = (e0 < E) ? e0 : (long long)(E - 1);
            long long c1 = (e1 < E) ? e1 : (long long)(E - 1);
            long long s0 = edge[c0], t0 = edge[(long long)E + c0];
            long long s1 = edge[c1], t1 = edge[(long long)E + c1];
            build_A(z + s0 * D_IN, z + t0 * D_IN, hi, A);
            build_A(z + s1 * D_IN, z + t1 * D_IN, hi, A + 8);
        }

        float part0[8] = {0.f,0.f,0.f,0.f,0.f,0.f,0.f,0.f};
        float part1[8] = {0.f,0.f,0.f,0.f,0.f,0.f,0.f,0.f};

        for (int nt = 0; nt < 16; ++nt) {
            const int n = nt * 16 + m;                 // hidden unit for this lane
            const char* rowp = smem + n * PITCH_B + hi * 16;
            v8f acc0 = {};
            v8f acc1 = {};
            #pragma unroll
            for (int kt = 0; kt < 8; ++kt) {
                // B 32x16 f16: lane holds column N=m, K = hi*8 + {0..7}, 16 + hi*8 + {0..7}
                const v8h* bp = (const v8h*)(rowp + kt * 64);
                v8h blo = bp[0];
                v8h bhi2 = bp[2];
                v16h B;
                #pragma unroll
                for (int i = 0; i < 8; ++i) { B[i] = blo[i]; B[8 + i] = bhi2[i]; }
                acc0 = __builtin_amdgcn_wmma_f32_16x16x32_f16(
                         false, A[kt],     false, B, (short)0, acc0, false, false);
                acc1 = __builtin_amdgcn_wmma_f32_16x16x32_f16(
                         false, A[8 + kt], false, B, (short)0, acc1, false, false);
            }
            // epilogue: bias + ELU + fused second layer (dot with W2)
            const float b1v = b1[n];
            const float w2v = W2[n];
            #pragma unroll
            for (int v = 0; v < 8; ++v) {
                float h0 = acc0[v] + b1v;
                float h1 = acc1[v] + b1v;
                float a0 = (h0 > 0.0f) ? h0 : (__expf(h0) - 1.0f);
                float a1 = (h1 > 0.0f) ? h1 : (__expf(h1) - 1.0f);
                part0[v] += a0 * w2v;
                part1[v] += a1 * w2v;
            }
        }

        // reduce each edge's partial across the 16 lanes of its half-wave
        #pragma unroll
        for (int v = 0; v < 8; ++v) {
            float r0 = part0[v];
            float r1 = part1[v];
            r0 += __shfl_xor(r0, 1, 32);  r1 += __shfl_xor(r1, 1, 32);
            r0 += __shfl_xor(r0, 2, 32);  r1 += __shfl_xor(r1, 2, 32);
            r0 += __shfl_xor(r0, 4, 32);  r1 += __shfl_xor(r1, 4, 32);
            r0 += __shfl_xor(r0, 8, 32);  r1 += __shfl_xor(r1, 8, 32);
            part0[v] = r0;
            part1[v] = r1;
        }
        if (m == 0) {                                  // lanes 0 and 16 write
            long long eb0 = ebase + hi * 8;
            long long eb1 = ebase + 16 + hi * 8;
            #pragma unroll
            for (int v = 0; v < 8; ++v) {
                long long e0 = eb0 + v;
                long long e1 = eb1 + v;
                if (e0 < E) out[e0] = part0[v] + b2v;
                if (e1 < E) out[e1] = part1[v] + b2v;
            }
        }
    }
}

extern "C" void kernel_launch(void* const* d_in, const int* in_sizes, int n_in,
                              void* d_out, int out_size, void* d_ws, size_t ws_size,
                              hipStream_t stream) {
    const float*     z    = (const float*)d_in[0];
    const long long* edge = (const long long*)d_in[1];
    const float*     W1   = (const float*)d_in[2];
    const float*     b1   = (const float*)d_in[3];
    const float*     W2   = (const float*)d_in[4];
    const float*     b2   = (const float*)d_in[5];
    float* out = (float*)d_out;

    const int E = in_sizes[1] / 2;          // edge is [2, E] int64
    const int ntiles = (E + 31) / 32;       // 32-edge tiles
    _Float16* w1h = (_Float16*)d_ws;

    // allow >64KB dynamic LDS (host-side attribute, capture-safe)
    (void)hipFuncSetAttribute((const void*)edge_mlp_kernel,
                              hipFuncAttributeMaxDynamicSharedMemorySize, LDS_BYTES);

    cvt_w1_kernel<<<256, 256, 0, stream>>>(W1, w1h);
    edge_mlp_kernel<<<512, 256, LDS_BYTES, stream>>>(z, edge, w1h, b1, W2, b2,
                                                     out, E, ntiles);
}
